// RNN_8057358648043
// MI455X (gfx1250) — matrix operand order repaired and checked by reading, not compiled, not run on hardware
//
#include <hip/hip_runtime.h>
#include <hip/hip_bf16.h>
#include <math.h>

// MI455X fused RNN: 16 independent batch-slab workgroups (B is the only
// dependency-free axis), weights resident in VGPRs as WMMA B-fragments,
// state in LDS, v_wmma_f32_16x16x32_f16 for all matmuls, hardware v_tanh_f32,
// 2 barriers per serial step, h_old fragments loaded once per step and reused,
// all A-fragments batched ahead of the WMMA chains, dual-accumulator chains.

typedef __attribute__((ext_vector_type(16))) _Float16 v16h;
typedef __attribute__((ext_vector_type(8)))  _Float16 v8h;
typedef __attribute__((ext_vector_type(8)))  float    v8f;
typedef __attribute__((ext_vector_type(4)))  float    v4f;

#define T_DIM 1024
#define B_DIM 256
#define D_DIM 64
#define H_DIM 128

__device__ __forceinline__ v8f wmma_f16(v16h a, v16h b, v8f c) {
  // D = A(16x32 f16) * B(32x16 f16) + C(16x16 f32)
  return __builtin_amdgcn_wmma_f32_16x16x32_f16(false, a, false, b,
                                                (short)0, c, false, false);
}

// Branch-free tanh: CDNA5 V_TANH_F32 (confirmed lowering), fallbacks kept.
__device__ __forceinline__ float fast_tanh(float x) {
#if __has_builtin(__builtin_amdgcn_tanhf)
  return __builtin_amdgcn_tanhf(x);
#elif __has_builtin(__builtin_amdgcn_tanh_f32)
  return __builtin_amdgcn_tanh_f32(x);
#else
  const float k2log2e = 2.885390081777926815f;  // 2*log2(e)
#if __has_builtin(__builtin_amdgcn_exp2f)
  float e = __builtin_amdgcn_exp2f(x * k2log2e);
#else
  float e = exp2f(x * k2log2e);
#endif
#if __has_builtin(__builtin_amdgcn_rcpf)
  return 1.0f - 2.0f * __builtin_amdgcn_rcpf(e + 1.0f);
#else
  return 1.0f - 2.0f / (e + 1.0f);
#endif
#endif
}

// B-fragment (32x16, per ISA f16 B layout): lane L holds column n,
// halves 0..7 = W[n][kb..kb+7], halves 8..15 = W[n][kb+16..kb+23],
// kb = k0 - (lane>>4)*8, k0 already includes (lane>>4)*16.
__device__ __forceinline__ v16h load_bfrag(const float* __restrict__ W, int ld,
                                           int n, int k0) {
  v16h b;
  const float* p = W + (size_t)n * ld + (k0 - ((k0 & 16) >> 1));
#pragma unroll
  for (int i = 0; i < 4; ++i) {
    v4f f = *(const v4f*)(p + 4 * i + (i >= 2 ? 8 : 0));
#pragma unroll
    for (int j = 0; j < 4; ++j) b[4 * i + j] = (_Float16)f[j];
  }
  return b;
}

// A-fragment (16x32 f16) from row-major f16 LDS tile (16 x ld):
// lane holds row m = lane&15; halves 0..7 = cols kb..kb+7,
// halves 8..15 = cols kb+16..kb+23, kb = ktile*32 + (lane>>4)*8.
__device__ __forceinline__ v16h load_afrag(const _Float16* buf, int ld,
                                           int m, int kb) {
  v8h lo = *(const v8h*)(buf + m * ld + kb);
  v8h hi = *(const v8h*)(buf + m * ld + kb + 16);
  v16h a;
#pragma unroll
  for (int i = 0; i < 8; ++i) { a[i] = lo[i]; a[8 + i] = hi[i]; }
  return a;
}

// tanh + f16-convert + scatter-store a 16x16 f32 C-tile column into LDS.
__device__ __forceinline__ void store_tile_tanh(_Float16* buf, int ld, int lk,
                                                int ncol, v8f acc) {
#pragma unroll
  for (int r = 0; r < 8; ++r)
    buf[(r + 8 * lk) * ld + ncol] = (_Float16)fast_tanh(acc[r]);
}

__global__ __launch_bounds__(256, 1) void rnn_fused_kernel(
    const float* __restrict__ x,
    const float* __restrict__ Wx,  const float* __restrict__ bx,
    const float* __restrict__ Wih, const float* __restrict__ bih,
    const float* __restrict__ Whh, const float* __restrict__ bhh,
    const float* __restrict__ Wh,  const float* __restrict__ bh,
    const float* __restrict__ Wg,  const float* __restrict__ bg,
    float* __restrict__ y) {
  __shared__ _Float16 xbuf[2][16 * D_DIM];     // double-buffered x_t slab, f16
  __shared__ _Float16 fbuf[16 * H_DIM];        // feat_t slab
  __shared__ _Float16 gbuf[16 * H_DIM];        // tanh(h Wh^T + bh)
  __shared__ _Float16 hbuf[2][16 * H_DIM];     // double-buffered hidden state

  const int tid  = threadIdx.x;
  const int lane = tid & 31;
  const int wv   = tid >> 5;          // wave 0..7: owns output cols [16wv,16wv+16)
  const int ln   = lane & 15;
  const int lk   = lane >> 4;
  const int b0   = blockIdx.x * 16;   // this workgroup's batch slab

  const int ncol = wv * 16 + ln;      // lane's output column in H (0..127)

  // ---- preload ALL weight fragments into registers (loop-invariant) ----
  v16h Bx[2], Bih[4], Bhh[4], Bh[4], Bg[4];
#pragma unroll
  for (int kt = 0; kt < 2; ++kt) Bx[kt]  = load_bfrag(Wx,  D_DIM, ncol, kt * 32 + lk * 16);
#pragma unroll
  for (int kt = 0; kt < 4; ++kt) Bih[kt] = load_bfrag(Wih, H_DIM, ncol, kt * 32 + lk * 16);
#pragma unroll
  for (int kt = 0; kt < 4; ++kt) Bhh[kt] = load_bfrag(Whh, H_DIM, ncol, kt * 32 + lk * 16);
#pragma unroll
  for (int kt = 0; kt < 4; ++kt) Bh[kt]  = load_bfrag(Wh,  H_DIM, ncol, kt * 32 + lk * 16);
  if (wv < 4) {  // waves 0..3 cover y's D=64 output columns
#pragma unroll
    for (int kt = 0; kt < 4; ++kt) Bg[kt] = load_bfrag(Wg, H_DIM, ncol, kt * 32 + lk * 16);
  } else {
#pragma unroll
    for (int kt = 0; kt < 4; ++kt) Bg[kt] = Bx[0];  // dead, keeps frags defined
  }

  const float bias_f = bx[ncol];                     // feat bias
  const float bias_u = bih[ncol] + bhh[ncol];        // fused recurrence bias
  const float bias_g = bh[ncol];                     // mlp_h bias
  const float bias_y = (wv < 4) ? bg[ncol] : 0.0f;   // output bias

  // ---- h0 = 0; prime x_0 into xbuf[0] ----
  for (int i = tid; i < 16 * H_DIM; i += 256) hbuf[0][i] = (_Float16)0.0f;
  {
    const float* xs = x + (size_t)b0 * D_DIM;
    v4f v = *(const v4f*)(xs + tid * 4);
    _Float16* d = xbuf[0] + tid * 4;
    d[0] = (_Float16)v[0]; d[1] = (_Float16)v[1];
    d[2] = (_Float16)v[2]; d[3] = (_Float16)v[3];
  }
  __syncthreads();

  int p = 0;
  for (int t = 0; t < T_DIM; ++t) {
    // ---------- phase 1 ----------
    // Prefetch-stage x_{t+1} into the other x slot (overlaps the WMMAs below).
    {
      const int tn = (t + 1 < T_DIM) ? (t + 1) : t;
      const float* xs = x + ((size_t)tn * B_DIM + b0) * D_DIM;
      v4f v = *(const v4f*)(xs + tid * 4);
      _Float16* d = xbuf[(t + 1) & 1] + tid * 4;
      d[0] = (_Float16)v[0]; d[1] = (_Float16)v[1];
      d[2] = (_Float16)v[2]; d[3] = (_Float16)v[3];
    }

    // Batch-load ALL phase-1 A-fragments first (12x ds_load_b128 in flight):
    // h_old fragments are kept in registers and REUSED in phase 2.
    v16h ha[4], xa[2];
#pragma unroll
    for (int kt = 0; kt < 4; ++kt)
      ha[kt] = load_afrag(hbuf[p], H_DIM, ln, kt * 32 + lk * 8);
#pragma unroll
    for (int kt = 0; kt < 2; ++kt)
      xa[kt] = load_afrag(xbuf[t & 1], D_DIM, ln, kt * 32 + lk * 8);

    // feat_t = tanh(x_t @ Wx^T + bx)
    {
      v8f a0, a1;
#pragma unroll
      for (int r = 0; r < 8; ++r) { a0[r] = bias_f; a1[r] = 0.0f; }
      a0 = wmma_f16(xa[0], Bx[0], a0);
      a1 = wmma_f16(xa[1], Bx[1], a1);
      store_tile_tanh(fbuf, H_DIM, lk, ncol, a0 + a1);
    }
    // g = tanh(h_old @ Wh^T + bh)  (dual accumulator chain)
    {
      v8f a0, a1;
#pragma unroll
      for (int r = 0; r < 8; ++r) { a0[r] = bias_g; a1[r] = 0.0f; }
      a0 = wmma_f16(ha[0], Bh[0], a0);
      a1 = wmma_f16(ha[1], Bh[1], a1);
      a0 = wmma_f16(ha[2], Bh[2], a0);
      a1 = wmma_f16(ha[3], Bh[3], a1);
      store_tile_tanh(gbuf, H_DIM, lk, ncol, a0 + a1);
    }
    __syncthreads();  // fbuf, gbuf ready (x_{t+1} staging also fenced here)

    // ---------- phase 2 ----------
    // Batch-load feat fragments (needed by all waves).
    v16h fa[4];
#pragma unroll
    for (int kt = 0; kt < 4; ++kt)
      fa[kt] = load_afrag(fbuf, H_DIM, ln, kt * 32 + lk * 8);

    // y_t = g @ Wg^T + bg  (uses h BEFORE the update; waves 0..3 -> D=64 cols)
    if (wv < 4) {
      v16h ga[4];
#pragma unroll
      for (int kt = 0; kt < 4; ++kt)
        ga[kt] = load_afrag(gbuf, H_DIM, ln, kt * 32 + lk * 8);
      v8f a0, a1;
#pragma unroll
      for (int r = 0; r < 8; ++r) { a0[r] = bias_y; a1[r] = 0.0f; }
      a0 = wmma_f16(ga[0], Bg[0], a0);
      a1 = wmma_f16(ga[1], Bg[1], a1);
      a0 = wmma_f16(ga[2], Bg[2], a0);
      a1 = wmma_f16(ga[3], Bg[3], a1);
      v8f acc = a0 + a1;
      float* yo = y + ((size_t)t * B_DIM + b0) * D_DIM + ncol;
#pragma unroll
      for (int r = 0; r < 8; ++r) yo[(size_t)(r + 8 * lk) * D_DIM] = acc[r];
    }
    // h_new = tanh(feat @ Wih^T + h_old @ Whh^T + (bih+bhh))
    // h_old fragments (ha) reused from phase 1 — zero LDS traffic for them.
    {
      v8f a0, a1;
#pragma unroll
      for (int r = 0; r < 8; ++r) { a0[r] = bias_u; a1[r] = 0.0f; }
      a0 = wmma_f16(ha[0], Bhh[0], a0);
      a1 = wmma_f16(ha[1], Bhh[1], a1);
      a0 = wmma_f16(ha[2], Bhh[2], a0);
      a1 = wmma_f16(ha[3], Bhh[3], a1);
      a0 = wmma_f16(fa[0], Bih[0], a0);
      a1 = wmma_f16(fa[1], Bih[1], a1);
      a0 = wmma_f16(fa[2], Bih[2], a0);
      a1 = wmma_f16(fa[3], Bih[3], a1);
      store_tile_tanh(hbuf[p ^ 1], H_DIM, lk, ncol, a0 + a1);
    }
    __syncthreads();  // h_new visible; fbuf/gbuf/xbuf slot reusable next step
    p ^= 1;
  }
}

extern "C" void kernel_launch(void* const* d_in, const int* in_sizes, int n_in,
                              void* d_out, int out_size, void* d_ws, size_t ws_size,
                              hipStream_t stream) {
  const float* x   = (const float*)d_in[0];
  const float* Wx  = (const float*)d_in[1];
  const float* bx  = (const float*)d_in[2];
  const float* Wih = (const float*)d_in[3];
  const float* bih = (const float*)d_in[4];
  const float* Whh = (const float*)d_in[5];
  const float* bhh = (const float*)d_in[6];
  const float* Wh  = (const float*)d_in[7];
  const float* bh  = (const float*)d_in[8];
  const float* Wg  = (const float*)d_in[9];
  const float* bg  = (const float*)d_in[10];
  float* y = (float*)d_out;

  rnn_fused_kernel<<<dim3(B_DIM / 16), dim3(256), 0, stream>>>(
      x, Wx, bx, Wih, bih, Whh, bhh, Wh, bh, Wg, bg, y);

  (void)in_sizes; (void)n_in; (void)out_size; (void)d_ws; (void)ws_size;
}
